// MultiHeadAttention_16183436771460
// MI455X (gfx1250) — matrix operand order repaired and checked by reading, not compiled, not run on hardware
//
#include <hip/hip_runtime.h>
#include <hip/hip_bf16.h>

// ---------------------------------------------------------------------------
// MI455X (gfx1250, wave32) Multi-Head Attention forward.
//   Pass 1: convert activations + weights to bf16 (one-shot, L2-resident after)
//   Pass 2: Q/K/V projections, bf16 WMMA, f32 accumulate (Q pre-scaled 1/8)
//   Pass 3: causal flash attention, 32 query rows per wave (2 tiles share K/V)
//   Pass 4: output projection, f32 result
// All matrix math on v_wmma_f32_16x16x32_bf16.
// ---------------------------------------------------------------------------

#define D_MODEL   1024
#define NUM_HEADS 16
#define DK        64
#define BATCH     4
#define SEQ       2048
#define M_TOK     (BATCH * SEQ)   // 8192 tokens

typedef __attribute__((ext_vector_type(16))) __bf16 v16bf;
typedef __attribute__((ext_vector_type(8)))  float  v8f;

union FragBF { v16bf v; unsigned u[8]; };
union BF2U   { __bf16 b; unsigned short u; };

__device__ __forceinline__ unsigned short f2bf(float f) {
  BF2U c; c.b = (__bf16)f;                 // hardware v_cvt path
  return c.u;
}
__device__ __forceinline__ unsigned pack2(float a, float b) {
  return (unsigned)f2bf(a) | ((unsigned)f2bf(b) << 16);
}
__device__ __forceinline__ v8f v8f_zero() {
  v8f z = {0.f, 0.f, 0.f, 0.f, 0.f, 0.f, 0.f, 0.f};
  return z;
}
__device__ __forceinline__ v8f wmma_bf16(v16bf a, v16bf b, v8f c) {
  // (neg_a, A, neg_b, B, c_mod, C, reuse_a, reuse_b)
  return __builtin_amdgcn_wmma_f32_16x16x32_bf16(false, a, false, b, (short)0, c,
                                                 false, false);
}

// --- Fragment loaders (ISA 7.12.2 layouts, wave32) --------------------------
// A (16x32 bf16, MxK): lane L: m=L&15, half=L>>4;
//   VGPR0..3 = K[8h..8h+7], VGPR4..7 = K[16+8h..16+8h+7]  (two B128 loads)
__device__ __forceinline__ v16bf load_a_bf16(const unsigned short* A, int ld,
                                             int mbase, int kbase, int lane) {
  const int m = lane & 15, half = lane >> 4;
  const unsigned short* p = A + (size_t)(mbase + m) * ld + kbase + 8 * half;
  uint4 c1 = *(const uint4*)(p);
  uint4 c2 = *(const uint4*)(p + 16);
  FragBF f;
  f.u[0]=c1.x; f.u[1]=c1.y; f.u[2]=c1.z; f.u[3]=c1.w;
  f.u[4]=c2.x; f.u[5]=c2.y; f.u[6]=c2.z; f.u[7]=c2.w;
  return f.v;
}
// B (32x16, KxN) with B[k][n] = Wm[n][k] (row-major Wm): lane L: n=L&15,
//   half=L>>4; VGPR0..3 = K[8h..8h+7], VGPR4..7 = K[16+8h..16+8h+7]
__device__ __forceinline__ v16bf load_bT_bf16(const unsigned short* Wm, int ld,
                                              int nbase, int kbase, int lane) {
  const int n = lane & 15, half = lane >> 4;
  const unsigned short* p = Wm + (size_t)(nbase + n) * ld + kbase + 8 * half;
  uint4 c1 = *(const uint4*)(p);
  uint4 c2 = *(const uint4*)(p + 16);
  FragBF f;
  f.u[0]=c1.x; f.u[1]=c1.y; f.u[2]=c1.z; f.u[3]=c1.w;
  f.u[4]=c2.x; f.u[5]=c2.y; f.u[6]=c2.z; f.u[7]=c2.w;
  return f.v;
}
__device__ __forceinline__ v16bf load_a_lds(const unsigned short* lds, int lane) {
  const int m = lane & 15, half = lane >> 4;
  const unsigned short* p = lds + m * 32 + 8 * half;
  uint4 c1 = *(const uint4*)(p);
  uint4 c2 = *(const uint4*)(p + 16);
  FragBF f;
  f.u[0]=c1.x; f.u[1]=c1.y; f.u[2]=c1.z; f.u[3]=c1.w;
  f.u[4]=c2.x; f.u[5]=c2.y; f.u[6]=c2.z; f.u[7]=c2.w;
  return f.v;
}

// ---------------------------------------------------------------------------
// One-shot f32 -> bf16 conversion (vectorized 8 elems/thread).
// ---------------------------------------------------------------------------
__global__ __launch_bounds__(256)
void cvt_bf16_kernel(const float* __restrict__ src,
                     unsigned short* __restrict__ dst, int n) {
  const int i = (blockIdx.x * 256 + threadIdx.x) * 8;
  if (i < n) {
    float4 f0 = *(const float4*)(src + i);
    float4 f1 = *(const float4*)(src + i + 4);
    uint4 o;
    o.x = pack2(f0.x, f0.y); o.y = pack2(f0.z, f0.w);
    o.z = pack2(f1.x, f1.y); o.w = pack2(f1.z, f1.w);
    *(uint4*)(dst + i) = o;
  }
}

// ---------------------------------------------------------------------------
// y = X * W^T GEMM, M=8192 N=1024 K=1024, all-bf16 inputs, f32 accumulate.
// 256 threads = 8 waves; wave tile 64x32 (4x2 WMMA, 8 WMMA / 12 B128 loads);
// block tile 128x128; K-loop software double-buffered.
// MODE 0  : store bf16*0.125 into [B,H,S,64]  (Q, pre-scaled by 1/sqrt(dk))
// MODE 1  : store bf16 into [B,H,S,64]        (K)
// MODE 2  : store bf16 into [B,H,64,S]        (V transposed)
// MODE 3  : store f32 [M,N]                   (output projection)
// ---------------------------------------------------------------------------
template <int MODE>
__global__ __launch_bounds__(256)
void gemm_xwT_kernel(const unsigned short* __restrict__ A,
                     const unsigned short* __restrict__ Wb,
                     void* __restrict__ Out) {
  const int lane = threadIdx.x & 31;
  const int wave = threadIdx.x >> 5;
  const int wm = wave & 1;               // 2 waves in M
  const int wn = wave >> 1;              // 4 waves in N
  const int mb = blockIdx.x * 128 + wm * 64;
  const int nb = blockIdx.y * 128 + wn * 32;

  v8f acc[4][2];
#pragma unroll
  for (int i = 0; i < 4; ++i) { acc[i][0] = v8f_zero(); acc[i][1] = v8f_zero(); }

  v16bf af[4], bf[2], an[4], bn[2];
#pragma unroll
  for (int i = 0; i < 4; ++i) af[i] = load_a_bf16(A, D_MODEL, mb + 16 * i, 0, lane);
  bf[0] = load_bT_bf16(Wb, D_MODEL, nb, 0, lane);
  bf[1] = load_bT_bf16(Wb, D_MODEL, nb + 16, 0, lane);

  for (int kb = 0; kb < D_MODEL; kb += 32) {
    const int kn = kb + 32;
    if (kn < D_MODEL) {                  // prefetch next K-step into registers
#pragma unroll
      for (int i = 0; i < 4; ++i) an[i] = load_a_bf16(A, D_MODEL, mb + 16 * i, kn, lane);
      bn[0] = load_bT_bf16(Wb, D_MODEL, nb, kn, lane);
      bn[1] = load_bT_bf16(Wb, D_MODEL, nb + 16, kn, lane);
    }
#pragma unroll
    for (int i = 0; i < 4; ++i) {
      acc[i][0] = wmma_bf16(af[i], bf[0], acc[i][0]);
      acc[i][1] = wmma_bf16(af[i], bf[1], acc[i][1]);
    }
    if (kn < D_MODEL) {
#pragma unroll
      for (int i = 0; i < 4; ++i) af[i] = an[i];
      bf[0] = bn[0]; bf[1] = bn[1];
    }
  }

  // C layout: VGPR r -> row 8*(lane>>4)+r, col = lane&15 within a 16x16 tile
  const int noff = lane & 15;
  const int moff = (lane >> 4) * 8;
#pragma unroll
  for (int i = 0; i < 4; ++i) {
#pragma unroll
    for (int j = 0; j < 2; ++j) {
#pragma unroll
      for (int r = 0; r < 8; ++r) {
        const int row = mb + i * 16 + moff + r;   // token index (b*S + s)
        const int col = nb + j * 16 + noff;       // feature index (h*64 + dk)
        float vres = acc[i][j][r];
        if constexpr (MODE == 3) {
          ((float*)Out)[(size_t)row * D_MODEL + col] = vres;
        } else {
          if constexpr (MODE == 0) vres *= 0.125f;   // fold 1/sqrt(dk) into Q
          const int b = row >> 11, s = row & (SEQ - 1);
          const int h = col >> 6,  d = col & (DK - 1);
          const unsigned short hv = f2bf(vres);
          if constexpr (MODE == 2) {   // V transposed: [B,H,64,S]
            ((unsigned short*)Out)[((size_t)(b * NUM_HEADS + h) * DK + d) * SEQ + s] = hv;
          } else {                     // Q,K: [B,H,S,64]
            ((unsigned short*)Out)[((size_t)(b * NUM_HEADS + h) * SEQ + s) * DK + d] = hv;
          }
        }
      }
    }
  }
}

// ---------------------------------------------------------------------------
// Online-softmax update for one 16x32 score tile pair (s0: keys kb..kb+15,
// s1: keys kb+16..kb+31). Row sums kept as per-lane partials (corr is
// lane-uniform), reduced once at kernel end. P written to LDS for A-reload.
// ---------------------------------------------------------------------------
template <bool MASKED>
__device__ __forceinline__ void softmax_update(
    v8f& s0, v8f& s1, v8f (&o)[4], float (&mrow)[8], float (&lpart)[8],
    unsigned short* __restrict__ Pl, int rowbase, int kb, int lane) {
  const int half = lane >> 4, nn = lane & 15, rowoff = half * 8;
#pragma unroll
  for (int r = 0; r < 8; ++r) {
    float v0 = s0[r], v1 = s1[r];
    if constexpr (MASKED) {
      const int row = rowbase + rowoff + r;
      const int c0 = kb + nn;
      v0 = (c0      <= row) ? v0 : -3.0e38f;
      v1 = (c0 + 16 <= row) ? v1 : -3.0e38f;
    }
    float t = fmaxf(v0, v1);                 // row max over 16 lanes of half
    t = fmaxf(t, __shfl_xor(t, 1, 32));
    t = fmaxf(t, __shfl_xor(t, 2, 32));
    t = fmaxf(t, __shfl_xor(t, 4, 32));
    t = fmaxf(t, __shfl_xor(t, 8, 32));
    const float mnew = fmaxf(mrow[r], t);
    const float corr = __expf(mrow[r] - mnew);
    const float p0 = __expf(v0 - mnew);
    const float p1 = __expf(v1 - mnew);
    lpart[r] = lpart[r] * corr + (p0 + p1);  // per-lane partial row sum
    mrow[r] = mnew;
    o[0][r] = o[0][r] * corr; o[1][r] = o[1][r] * corr;
    o[2][r] = o[2][r] * corr; o[3][r] = o[3][r] * corr;
    Pl[(rowoff + r) * 32 + nn]      = f2bf(p0);
    Pl[(rowoff + r) * 32 + 16 + nn] = f2bf(p1);
  }
}

// One 32-key block against 32 query rows (two 16-row tiles sharing K/V frags).
template <bool MASKED>
__device__ __forceinline__ void attn_block32(
    int kb, int qbase, int lane,
    const unsigned short* __restrict__ Kh, const unsigned short* __restrict__ Vh,
    unsigned short* __restrict__ Plds,
    const v16bf (&qa)[2][2], v8f (&o)[2][4],
    float (&mrow)[2][8], float (&lpart)[2][8]) {
  // Shared K/V fragments for both q-tiles; issue early to overlap softmax VALU.
  v16bf kf[4], vf[4];
  kf[0] = load_bT_bf16(Kh, DK, kb,      0,  lane);
  kf[1] = load_bT_bf16(Kh, DK, kb,      32, lane);
  kf[2] = load_bT_bf16(Kh, DK, kb + 16, 0,  lane);
  kf[3] = load_bT_bf16(Kh, DK, kb + 16, 32, lane);
#pragma unroll
  for (int j = 0; j < 4; ++j) vf[j] = load_bT_bf16(Vh, SEQ, 16 * j, kb, lane);

#pragma unroll
  for (int t = 0; t < 2; ++t) {
    v8f s0 = v8f_zero(), s1 = v8f_zero();
    s0 = wmma_bf16(qa[t][0], kf[0], s0);
    s0 = wmma_bf16(qa[t][1], kf[1], s0);
    s1 = wmma_bf16(qa[t][0], kf[2], s1);
    s1 = wmma_bf16(qa[t][1], kf[3], s1);
    softmax_update<MASKED>(s0, s1, o[t], mrow[t], lpart[t],
                           Plds + t * 512, qbase + 16 * t, kb, lane);
  }
  asm volatile("s_wait_dscnt 0x0" ::: "memory");   // LDS in-order per wave

#pragma unroll
  for (int t = 0; t < 2; ++t) {
    const v16bf pa = load_a_lds(Plds + t * 512, lane);
#pragma unroll
    for (int j = 0; j < 4; ++j) o[t][j] = wmma_bf16(pa, vf[j], o[t][j]);
  }
}

// Flash attention, causal. One wave per 32 query rows of one (b,h).
// Single-wave workgroup -> barriers are NOPs, LDS ops in-order per wave.
__global__ __launch_bounds__(32)
void attn_kernel(const unsigned short* __restrict__ Q,
                 const unsigned short* __restrict__ K,
                 const unsigned short* __restrict__ Vt,
                 unsigned short* __restrict__ Ctx) {
  __shared__ __align__(16) unsigned short Plds[2 * 16 * 32];  // two P tiles

  const int lane  = threadIdx.x;
  const int bh    = blockIdx.x >> 6;            // 64 q-blocks per (b,h)
  const int qt    = blockIdx.x & 63;
  const int qbase = qt * 32;                    // 32-aligned
  const int b     = bh >> 4, h = bh & 15;

  const unsigned short* Qh = Q  + (size_t)bh * SEQ * DK;
  const unsigned short* Kh = K  + (size_t)bh * SEQ * DK;
  const unsigned short* Vh = Vt + (size_t)bh * DK * SEQ;   // [64][S]

  // Q fragments (Q already carries the 1/sqrt(dk) scale from the projection)
  v16bf qa[2][2];
  qa[0][0] = load_a_bf16(Qh, DK, qbase,      0,  lane);
  qa[0][1] = load_a_bf16(Qh, DK, qbase,      32, lane);
  qa[1][0] = load_a_bf16(Qh, DK, qbase + 16, 0,  lane);
  qa[1][1] = load_a_bf16(Qh, DK, qbase + 16, 32, lane);

  v8f o[2][4];
  float mrow[2][8], lpart[2][8];
#pragma unroll
  for (int t = 0; t < 2; ++t)
#pragma unroll
    for (int r = 0; r < 8; ++r) {
      mrow[t][r] = -3.0e38f; lpart[t][r] = 0.0f;
      if (r < 4) o[t][r] = v8f_zero();
    }

  const int nn = lane & 15;
  const int rowoff = (lane >> 4) * 8;

  // qbase is 32-aligned: all blocks kb < qbase are fully unmasked; exactly one
  // masked diagonal block at kb == qbase.
  for (int kb = 0; kb < qbase; kb += 32) {
    __builtin_prefetch(Kh + (size_t)(kb + 32 + nn) * DK, 0, 1);
    __builtin_prefetch(Vh + (size_t)nn * SEQ + kb + 32, 0, 1);
    attn_block32<false>(kb, qbase, lane, Kh, Vh, Plds, qa, o, mrow, lpart);
  }
  attn_block32<true>(qbase, qbase, lane, Kh, Vh, Plds, qa, o, mrow, lpart);

  // Finish deferred row-sum reduction, normalize, store ctx [B,S,D] bf16.
#pragma unroll
  for (int t = 0; t < 2; ++t) {
#pragma unroll
    for (int r = 0; r < 8; ++r) {
      float l = lpart[t][r];
      l += __shfl_xor(l, 1, 32);
      l += __shfl_xor(l, 2, 32);
      l += __shfl_xor(l, 4, 32);
      l += __shfl_xor(l, 8, 32);
      const float inv = 1.0f / l;
      const int tok = b * SEQ + qbase + 16 * t + rowoff + r;
      const size_t base = (size_t)tok * D_MODEL + h * DK + nn;
      Ctx[base + 0]  = f2bf(o[t][0][r] * inv);
      Ctx[base + 16] = f2bf(o[t][1][r] * inv);
      Ctx[base + 32] = f2bf(o[t][2][r] * inv);
      Ctx[base + 48] = f2bf(o[t][3][r] * inv);
    }
  }
}

// ---------------------------------------------------------------------------
extern "C" void kernel_launch(void* const* d_in, const int* in_sizes, int n_in,
                              void* d_out, int out_size, void* d_ws, size_t ws_size,
                              hipStream_t stream) {
  const float* query = (const float*)d_in[0];
  const float* key_i = (const float*)d_in[1];
  const float* value = (const float*)d_in[2];
  // d_in[3] = causal mask (int32 tril) -- applied analytically in attn_kernel
  const float* w_q = (const float*)d_in[4];
  const float* w_k = (const float*)d_in[5];
  const float* w_v = (const float*)d_in[6];
  const float* w_o = (const float*)d_in[7];

  const size_t actE = (size_t)M_TOK * D_MODEL;     // 8.39M elems
  const size_t wE   = (size_t)D_MODEL * D_MODEL;   // 1.05M elems

  // Workspace (bf16): 3 activations + 4 weights + Q + K + Vt + Ctx = 125.8 MB,
  // all L2-resident on the 192 MB global L2.
  unsigned short* p   = (unsigned short*)d_ws;
  unsigned short* Xq  = p; p += actE;
  unsigned short* Xk  = p; p += actE;
  unsigned short* Xv  = p; p += actE;
  unsigned short* Wqb = p; p += wE;
  unsigned short* Wkb = p; p += wE;
  unsigned short* Wvb = p; p += wE;
  unsigned short* Wob = p; p += wE;
  unsigned short* Qw  = p; p += actE;
  unsigned short* Kw  = p; p += actE;
  unsigned short* Vtw = p; p += actE;
  unsigned short* Ctx = p;

  const int actBlk = (int)(actE / (256 * 8));      // 4096
  const int wBlk   = (int)(wE / (256 * 8));        // 512
  cvt_bf16_kernel<<<actBlk, 256, 0, stream>>>(query, Xq, (int)actE);
  cvt_bf16_kernel<<<actBlk, 256, 0, stream>>>(key_i, Xk, (int)actE);
  cvt_bf16_kernel<<<actBlk, 256, 0, stream>>>(value, Xv, (int)actE);
  cvt_bf16_kernel<<<wBlk, 256, 0, stream>>>(w_q, Wqb, (int)wE);
  cvt_bf16_kernel<<<wBlk, 256, 0, stream>>>(w_k, Wkb, (int)wE);
  cvt_bf16_kernel<<<wBlk, 256, 0, stream>>>(w_v, Wvb, (int)wE);
  cvt_bf16_kernel<<<wBlk, 256, 0, stream>>>(w_o, Wob, (int)wE);

  const dim3 gg(M_TOK / 128, D_MODEL / 128);       // 64 x 8 blocks
  gemm_xwT_kernel<0><<<gg, 256, 0, stream>>>(Xq, Wqb, Qw);
  gemm_xwT_kernel<1><<<gg, 256, 0, stream>>>(Xk, Wkb, Kw);
  gemm_xwT_kernel<2><<<gg, 256, 0, stream>>>(Xv, Wvb, Vtw);

  attn_kernel<<<BATCH * NUM_HEADS * (SEQ / 32), 32, 0, stream>>>(Qw, Kw, Vtw, Ctx);

  gemm_xwT_kernel<3><<<gg, 256, 0, stream>>>(Ctx, Wob, (float*)d_out);
}
